// GCN_scheduling_67834713473214
// MI455X (gfx1250) — matrix operand order repaired and checked by reading, not compiled, not run on hardware
//
#include <hip/hip_runtime.h>
#include <math.h>

#define N_NODES 50000
#define E_EDGES 800000
#define EP      (E_EDGES + N_NODES)   // edges + self loops = 850000
#define F_IN    1284
#define KP      1312                  // F_IN padded to multiple of 32 (41*32)
#define KSTEPS  41                    // 40 full steps + 1 guarded tail step
#define H       128

typedef __attribute__((ext_vector_type(16))) __bf16 v16bf;
typedef __attribute__((ext_vector_type(8)))  __bf16 v8bf;
typedef __attribute__((ext_vector_type(8)))  float  v8f;
typedef __attribute__((ext_vector_type(4)))  int    v4i;

// ---- gfx1250 async global->LDS path (guarded: falls back to reg bounce) ----
#if defined(__gfx1250__) && __has_builtin(__builtin_amdgcn_global_load_async_to_lds_b128) && __has_builtin(__builtin_amdgcn_s_wait_asynccnt)
#define USE_ASYNC_LDS 1
#else
#define USE_ASYNC_LDS 0
#endif

#if USE_ASYNC_LDS
typedef __attribute__((address_space(1))) v4i* gptr_v4i;
typedef __attribute__((address_space(3))) v4i* lptr_v4i;
__device__ __forceinline__ void async_ld_b128(const void* g, void* l) {
    __builtin_amdgcn_global_load_async_to_lds_b128(
        (gptr_v4i)(v4i*)g, (lptr_v4i)(v4i*)l, 0, 0);
}
#endif
__device__ __forceinline__ void wait_async_lds() {
#if USE_ASYNC_LDS
    __builtin_amdgcn_s_wait_asynccnt(0);
#endif
}

// ---------- ordered-float encoding for u32 atomicMax (segment max) ----------
__device__ __forceinline__ unsigned f2ord(float f) {
    unsigned u = __float_as_uint(f);
    return (u & 0x80000000u) ? ~u : (u | 0x80000000u);
}
__device__ __forceinline__ float ord2f(unsigned o) {
    unsigned u = (o & 0x80000000u) ? (o ^ 0x80000000u) : ~o;
    return __uint_as_float(u);
}
__device__ __forceinline__ float wave_sum(float s) {
    #pragma unroll
    for (int off = 16; off; off >>= 1) s += __shfl_down(s, off, 32);
    return s;
}

// ---------- W1 (F_IN x H, fp32) -> w1t (H x KP, bf16, col-of-W1 contiguous in K)
__global__ void k_convert_w1(const float* __restrict__ W1, __bf16* __restrict__ w1t) {
    int idx = blockIdx.x * blockDim.x + threadIdx.x;
    if (idx >= H * KP) return;
    int n = idx / KP, k = idx % KP;
    w1t[idx] = (k < F_IN) ? (__bf16)W1[(size_t)k * H + n] : (__bf16)0.0f;
}

// ---------- scal[1] = dot(We1, att_e1); scal[2] = We2[0]*att_e2[0] ----------
__global__ void k_scalars(const float* We1, const float* att_e1,
                          const float* We2, const float* att_e2, float* scal) {
    __shared__ float red[128];
    int t = threadIdx.x;
    red[t] = We1[t] * att_e1[t];
    __syncthreads();
    #pragma unroll
    for (int s = 64; s > 0; s >>= 1) { if (t < s) red[t] += red[t + s]; __syncthreads(); }
    if (t == 0) { scal[1] = red[0]; scal[2] = We2[0] * att_e2[0]; }
}

// ---------- scal[0] += sum(edge_attr) ----------
__global__ void k_esum(const float* __restrict__ ea, float* scal) {
    int gid = blockIdx.x * blockDim.x + threadIdx.x;
    float s = 0.f;
    for (int i = gid; i < E_EDGES; i += gridDim.x * blockDim.x) s += ea[i];
    s = wave_sum(s);
    if ((threadIdx.x & 31) == 0) atomicAdd(&scal[0], s);
}

// ---------- h1 = x @ W1  via v_wmma_f32_16x16x32_bf16, double-buffered LDS ----
__global__ __launch_bounds__(256) void k_gemm1(const float* __restrict__ x,
                                               const __bf16* __restrict__ w1t,
                                               float* __restrict__ h1) {
    __shared__ __bf16 a_lds[2][128 * 32];
    __shared__ __bf16 b_lds[2][128 * 32];
    const int tid  = threadIdx.x;
    const int wave = tid >> 5, lane = tid & 31;
    const int row0 = blockIdx.x * 128;
    const int r     = tid >> 1;            // staging row / col 0..127
    const int halfk = (tid & 1) * 16;      // 0 or 16
    const int rowc  = (row0 + r < N_NODES) ? (row0 + r) : (N_NODES - 1);
    const float*  xrow = x   + (size_t)rowc * F_IN;
    const __bf16* brow = w1t + (size_t)r * KP;

    v8f acc[8];
    #pragma unroll
    for (int i = 0; i < 8; ++i) acc[i] = (v8f){0.f,0.f,0.f,0.f,0.f,0.f,0.f,0.f};

    // convert 16 staged fp32 -> two packed v8bf LDS stores (2x ds_store_b128)
    auto store_a = [&](int buf, const float* tv) {
        __bf16* ad = &a_lds[buf][r * 32 + halfk];
        v8bf o0, o1;
        #pragma unroll
        for (int j = 0; j < 8; ++j) { o0[j] = (__bf16)tv[j]; o1[j] = (__bf16)tv[8 + j]; }
        *(v8bf*)ad       = o0;
        *(v8bf*)(ad + 8) = o1;
    };
    auto compute_step = [&](const __bf16* ab, const __bf16* bb) {
        const int m = lane & 15, hh = lane >> 4;
        const __bf16* ap = ab + (wave * 16 + m) * 32 + 16 * hh;
        v8bf alo = *(const v8bf*)ap;          // K = 16h .. 16h+7
        v8bf ahi = *(const v8bf*)(ap + 8);    // K = 16h+8 .. 16h+15
        v16bf af = __builtin_shufflevector(alo, ahi,
                     0,1,2,3,4,5,6,7,8,9,10,11,12,13,14,15);
        #pragma unroll
        for (int ct = 0; ct < 8; ++ct) {
            const __bf16* bp = bb + (ct * 16 + m) * 32 + 16 * hh;
            v16bf bf_ = *(const v16bf*)bp;
            acc[ct] = __builtin_amdgcn_wmma_f32_16x16x32_bf16(
                false, af, false, bf_, (short)0, acc[ct], false, false);
        }
    };

    // ---- prologue: stage step 0 into buffer 0 ----
    {
        const float4* xs4 = (const float4*)(xrow + halfk);
        float4 a0 = xs4[0], a1 = xs4[1], a2 = xs4[2], a3 = xs4[3];
        float tv[16] = {a0.x,a0.y,a0.z,a0.w, a1.x,a1.y,a1.z,a1.w,
                        a2.x,a2.y,a2.z,a2.w, a3.x,a3.y,a3.z,a3.w};
        store_a(0, tv);
        const __bf16* bs = brow + halfk;
#if USE_ASYNC_LDS
        async_ld_b128(bs,     &b_lds[0][r * 32 + halfk]);
        async_ld_b128(bs + 8, &b_lds[0][r * 32 + halfk + 8]);
#else
        *(v8bf*)&b_lds[0][r * 32 + halfk]     = *(const v8bf*)bs;
        *(v8bf*)&b_lds[0][r * 32 + halfk + 8] = *(const v8bf*)(bs + 8);
#endif
    }
    wait_async_lds();
    __syncthreads();

    // ---- main pipeline: compute step i while staging step i+1 ----
    for (int i = 0; i < KSTEPS - 1; ++i) {
        const int cur = i & 1, nxt = cur ^ 1;
        const int k0n = (i + 1) * 32;
        float tv[16];
        v8bf nb0, nb1;
        // issue next-step loads before compute (latency hidden under WMMA)
        if (i < KSTEPS - 2) {                 // full 32-wide A slice
            const float4* xs4 = (const float4*)(xrow + k0n + halfk);
            float4 a0 = xs4[0], a1 = xs4[1], a2 = xs4[2], a3 = xs4[3];
            tv[0]=a0.x; tv[1]=a0.y; tv[2]=a0.z; tv[3]=a0.w;
            tv[4]=a1.x; tv[5]=a1.y; tv[6]=a1.z; tv[7]=a1.w;
            tv[8]=a2.x; tv[9]=a2.y; tv[10]=a2.z; tv[11]=a2.w;
            tv[12]=a3.x; tv[13]=a3.y; tv[14]=a3.z; tv[15]=a3.w;
        } else {                              // guarded K tail (1280..1283)
            #pragma unroll
            for (int j = 0; j < 16; ++j) {
                int k = k0n + halfk + j;
                tv[j] = (k < F_IN) ? xrow[k] : 0.0f;
            }
        }
        {
            const __bf16* bs = brow + k0n + halfk;   // w1t zero-padded: always full
#if USE_ASYNC_LDS
            async_ld_b128(bs,     &b_lds[nxt][r * 32 + halfk]);
            async_ld_b128(bs + 8, &b_lds[nxt][r * 32 + halfk + 8]);
#else
            nb0 = *(const v8bf*)bs; nb1 = *(const v8bf*)(bs + 8);
#endif
        }

        compute_step(&a_lds[cur][0], &b_lds[cur][0]);

        store_a(nxt, tv);
#if !USE_ASYNC_LDS
        *(v8bf*)&b_lds[nxt][r * 32 + halfk]     = nb0;
        *(v8bf*)&b_lds[nxt][r * 32 + halfk + 8] = nb1;
#endif
        wait_async_lds();
        __syncthreads();
    }

    // ---- tail compute: step 40 lives in buffer (40 & 1) == 0 ----
    compute_step(&a_lds[0][0], &b_lds[0][0]);

    // ---- store C: vgpr rr -> row (rr + 8*(lane>=16)), col = lane&15 ----
    const int nn = lane & 15, mb = (lane >> 4) * 8;
    #pragma unroll
    for (int ct = 0; ct < 8; ++ct) {
        #pragma unroll
        for (int rr = 0; rr < 8; ++rr) {
            int row = row0 + wave * 16 + mb + rr;
            if (row < N_NODES) h1[(size_t)row * H + ct * 16 + nn] = acc[ct][rr];
        }
    }
}

// ---------- per-node attention logits: alpha_src/dst = h1 . att ----------
__global__ void k_alpha1(const float* __restrict__ h1, const float* __restrict__ a_src,
                         const float* __restrict__ a_dst, float* as_, float* ad_) {
    int wid  = (blockIdx.x * blockDim.x + threadIdx.x) >> 5;
    int lane = threadIdx.x & 31;
    if (wid >= N_NODES) return;
    const float* hp = h1 + (size_t)wid * H;
    float s0 = 0.f, s1 = 0.f;
    #pragma unroll
    for (int i = lane; i < H; i += 32) { float hv = hp[i]; s0 += hv * a_src[i]; s1 += hv * a_dst[i]; }
    s0 = wave_sum(s0); s1 = wave_sum(s1);
    if (lane == 0) { as_[wid] = s0; ad_[wid] = s1; }
}

__device__ __forceinline__ void edge_sd(int e, const int* ei, int& s, int& d) {
    if (e < E_EDGES) { s = ei[e]; d = ei[E_EDGES + e]; }
    else             { s = d = e - E_EDGES; }
}

// ---------- layer-1 softmax pass 1: alpha + segment max ----------
__global__ void k_edge_max1(const int* __restrict__ ei, const float* __restrict__ eattr,
                            const float* scal, const float* __restrict__ as_,
                            const float* __restrict__ ad_, float* __restrict__ ealpha,
                            unsigned* __restrict__ segmax) {
    int e = blockIdx.x * blockDim.x + threadIdx.x;
    if (e >= EP) return;
    int s, d; edge_sd(e, ei, s, d);
    float ev = (e < E_EDGES) ? eattr[e] : scal[0] * (1.0f / E_EDGES);
    float a = as_[s] + ad_[d] + ev * scal[1];
    a = a > 0.f ? a : 0.2f * a;
    ealpha[e] = a;
    atomicMax(segmax + d, f2ord(a));
}

// ---------- softmax pass 2: exp + segment sum ----------
__global__ void k_edge_exp(const int* __restrict__ ei, float* __restrict__ ealpha,
                           const unsigned* __restrict__ segmax, float* __restrict__ denom) {
    int e = blockIdx.x * blockDim.x + threadIdx.x;
    if (e >= EP) return;
    int s, d; edge_sd(e, ei, s, d);
    float v = __expf(ealpha[e] - ord2f(segmax[d]));
    ealpha[e] = v;
    atomicAdd(denom + d, v);
}

// ---------- layer-1 scatter: out1[d] += coef * h1[s]  (wave per edge) ----------
__global__ void k_scatter1(const int* __restrict__ ei, const float* __restrict__ ealpha,
                           const float* __restrict__ denom, const float* __restrict__ h1,
                           float* __restrict__ out1) {
    int e    = (blockIdx.x * blockDim.x + threadIdx.x) >> 5;
    int lane = threadIdx.x & 31;
    if (e >= EP) return;
    int s, d; edge_sd(e, ei, s, d);
    float coef = ealpha[e] / (denom[d] + 1e-16f);
    const float* hp = h1 + (size_t)s * H;
    float* op = out1 + (size_t)d * H;
    #pragma unroll
    for (int i = lane; i < H; i += 32) atomicAdd(op + i, coef * hp[i]);
}

// ---------- node: x2 = elu(out1 + b1);  h2 = x2 . W2 ----------
__global__ void k_node2(const float* __restrict__ out1, const float* __restrict__ b1,
                        const float* __restrict__ W2, float* __restrict__ h2) {
    int wid  = (blockIdx.x * blockDim.x + threadIdx.x) >> 5;
    int lane = threadIdx.x & 31;
    if (wid >= N_NODES) return;
    const float* op = out1 + (size_t)wid * H;
    float s = 0.f;
    #pragma unroll
    for (int i = lane; i < H; i += 32) {
        float x1 = op[i] + b1[i];
        float x2 = x1 > 0.f ? x1 : (__expf(x1) - 1.0f);   // ELU
        s += x2 * W2[i];
    }
    s = wave_sum(s);
    if (lane == 0) h2[wid] = s;
}

// ---------- layer-2 softmax pass 1 ----------
__global__ void k_edge_max2(const int* __restrict__ ei, const float* __restrict__ eattr,
                            const float* scal, const float* __restrict__ h2,
                            const float* as2, const float* ad2,
                            float* __restrict__ ealpha, unsigned* __restrict__ segmax) {
    int e = blockIdx.x * blockDim.x + threadIdx.x;
    if (e >= EP) return;
    int s, d; edge_sd(e, ei, s, d);
    float ev = (e < E_EDGES) ? eattr[e] : scal[0] * (1.0f / E_EDGES);
    float a = h2[s] * as2[0] + h2[d] * ad2[0] + ev * scal[2];
    a = a > 0.f ? a : 0.2f * a;
    ealpha[e] = a;
    atomicMax(segmax + d, f2ord(a));
}

// ---------- layer-2 scatter (scalar) ----------
__global__ void k_scatter2(const int* __restrict__ ei, const float* __restrict__ ealpha,
                           const float* __restrict__ denom, const float* __restrict__ h2,
                           float* __restrict__ out2) {
    int e = blockIdx.x * blockDim.x + threadIdx.x;
    if (e >= EP) return;
    int s, d; edge_sd(e, ei, s, d);
    float coef = ealpha[e] / (denom[d] + 1e-16f);
    atomicAdd(out2 + d, coef * h2[s]);
}

// ---------- final leaky relu ----------
__global__ void k_final(const float* __restrict__ out2, const float* b2, float* __restrict__ out) {
    int n = blockIdx.x * blockDim.x + threadIdx.x;
    if (n >= N_NODES) return;
    float v = out2[n] + b2[0];
    out[n] = v > 0.f ? v : 0.01f * v;
}

extern "C" void kernel_launch(void* const* d_in, const int* in_sizes, int n_in,
                              void* d_out, int out_size, void* d_ws, size_t ws_size,
                              hipStream_t stream) {
    const float* x        = (const float*)d_in[0];
    const int*   ei       = (const int*)  d_in[1];
    const float* eattr    = (const float*)d_in[2];
    const float* W1       = (const float*)d_in[3];
    const float* att_src1 = (const float*)d_in[4];
    const float* att_dst1 = (const float*)d_in[5];
    const float* We1      = (const float*)d_in[6];
    const float* att_e1   = (const float*)d_in[7];
    const float* b1       = (const float*)d_in[8];
    const float* W2       = (const float*)d_in[9];
    const float* att_src2 = (const float*)d_in[10];
    const float* att_dst2 = (const float*)d_in[11];
    const float* We2      = (const float*)d_in[12];
    const float* att_e2   = (const float*)d_in[13];
    const float* b2       = (const float*)d_in[14];
    float* out = (float*)d_out;

    // workspace carve-out (256B aligned)
    char* w = (char*)d_ws;
    auto alloc = [&](size_t bytes) { char* p = w; w += (bytes + 255) & ~(size_t)255; return p; };
    __bf16*   w1t     = (__bf16*)  alloc((size_t)KP * H * 2);
    float*    h1      = (float*)   alloc((size_t)N_NODES * H * 4);
    float*    as1     = (float*)   alloc((size_t)N_NODES * 4);
    float*    ad1     = (float*)   alloc((size_t)N_NODES * 4);
    unsigned* segmax1 = (unsigned*)alloc((size_t)N_NODES * 4);
    float*    denom1  = (float*)   alloc((size_t)N_NODES * 4);
    float*    ea1     = (float*)   alloc((size_t)EP * 4);
    float*    out1    = (float*)   alloc((size_t)N_NODES * H * 4);
    float*    h2      = (float*)   alloc((size_t)N_NODES * 4);
    unsigned* segmax2 = (unsigned*)alloc((size_t)N_NODES * 4);
    float*    denom2  = (float*)   alloc((size_t)N_NODES * 4);
    float*    ea2     = (float*)   alloc((size_t)EP * 4);
    float*    out2    = (float*)   alloc((size_t)N_NODES * 4);
    float*    scal    = (float*)   alloc(16);   // [0]=esum [1]=ce1 [2]=ce2

    // re-init accumulators every call (graph replays must be deterministic)
    (void)hipMemsetAsync(segmax1, 0, (size_t)N_NODES * 4, stream);  // 0 = u32 min, below ord(-inf)
    (void)hipMemsetAsync(denom1,  0, (size_t)N_NODES * 4, stream);
    (void)hipMemsetAsync(out1,    0, (size_t)N_NODES * H * 4, stream);
    (void)hipMemsetAsync(segmax2, 0, (size_t)N_NODES * 4, stream);
    (void)hipMemsetAsync(denom2,  0, (size_t)N_NODES * 4, stream);
    (void)hipMemsetAsync(out2,    0, (size_t)N_NODES * 4, stream);
    (void)hipMemsetAsync(scal,    0, 16, stream);

    const int eb = (EP + 255) / 256;

    k_convert_w1<<<(KP * H + 255) / 256, 256, 0, stream>>>(W1, w1t);
    k_scalars   <<<1, 128, 0, stream>>>(We1, att_e1, We2, att_e2, scal);
    k_esum      <<<256, 256, 0, stream>>>(eattr, scal);
    k_gemm1     <<<(N_NODES + 127) / 128, 256, 0, stream>>>(x, w1t, h1);
    k_alpha1    <<<(N_NODES + 7) / 8, 256, 0, stream>>>(h1, att_src1, att_dst1, as1, ad1);
    k_edge_max1 <<<eb, 256, 0, stream>>>(ei, eattr, scal, as1, ad1, ea1, segmax1);
    k_edge_exp  <<<eb, 256, 0, stream>>>(ei, ea1, segmax1, denom1);
    k_scatter1  <<<(EP + 7) / 8, 256, 0, stream>>>(ei, ea1, denom1, h1, out1);
    k_node2     <<<(N_NODES + 7) / 8, 256, 0, stream>>>(out1, b1, W2, h2);
    k_edge_max2 <<<eb, 256, 0, stream>>>(ei, eattr, scal, h2, att_src2, att_dst2, ea2, segmax2);
    k_edge_exp  <<<eb, 256, 0, stream>>>(ei, ea2, segmax2, denom2);
    k_scatter2  <<<eb, 256, 0, stream>>>(ei, ea2, denom2, h2, out2);
    k_final     <<<(N_NODES + 255) / 256, 256, 0, stream>>>(out2, b2, out);
}
